// TransformerLayer_30932354466187
// MI455X (gfx1250) — matrix operand order repaired and checked by reading, not compiled, not run on hardware
//
#include <hip/hip_runtime.h>

// Problem constants (B,S,D,H,DH,FF from reference)
#define Bc  2
#define Sc  2048
#define Dc  1024
#define Hc  16
#define DHc 64
#define FFc 4096
#define BSc (Bc * Sc)

typedef __attribute__((ext_vector_type(16))) __bf16 v16bf;
typedef __attribute__((ext_vector_type(8)))  float  v8f;
typedef __attribute__((ext_vector_type(4)))  int    v4i;

union Frag { v16bf v; unsigned int u[8]; };

__device__ __forceinline__ unsigned short f2bf(float f) {
    unsigned int u = __float_as_uint(f);
    u += 0x7FFFu + ((u >> 16) & 1u);   // round-to-nearest-even
    return (unsigned short)(u >> 16);
}
__device__ __forceinline__ float bf2f(unsigned short h) {
    return __uint_as_float(((unsigned int)h) << 16);
}

// ---- CDNA5 async global->LDS copy (ASYNCcnt-tracked), with fallback ----------
#if __has_builtin(__builtin_amdgcn_global_load_async_to_lds_b128)
#define HAVE_ASYNC_LDS 1
__device__ __forceinline__ void async_ld_b128(const void* g, void* l) {
    __builtin_amdgcn_global_load_async_to_lds_b128((v4i*)g, (v4i*)l, 0, 0);
}
__device__ __forceinline__ void async_wait0() {
#if __has_builtin(__builtin_amdgcn_s_wait_asynccnt)
    __builtin_amdgcn_s_wait_asynccnt(0);
#else
    asm volatile("s_wait_asynccnt 0" ::: "memory");
#endif
}
#endif

// ---------------------------------------------------------------- cvt f32->bf16
__global__ __launch_bounds__(256)
void cvt_kernel(const float* __restrict__ in, unsigned short* __restrict__ out, int n) {
    int i = blockIdx.x * 256 + threadIdx.x;
    if (i < n) out[i] = f2bf(in[i]);
}

// ---------------------------------------------------------------- RMSNorm -> bf16
__global__ __launch_bounds__(256)
void rmsnorm_kernel(const float* __restrict__ x, const float* __restrict__ g,
                    unsigned short* __restrict__ out) {
    __shared__ float red[256];
    int row = blockIdx.x;
    const float* xr = x + (size_t)row * Dc;
    float ss = 0.f;
    for (int i = threadIdx.x; i < Dc; i += 256) { float v = xr[i]; ss += v * v; }
    red[threadIdx.x] = ss;
    __syncthreads();
    for (int s = 128; s > 0; s >>= 1) {
        if (threadIdx.x < s) red[threadIdx.x] += red[threadIdx.x + s];
        __syncthreads();
    }
    float inv = rsqrtf(red[0] * (1.0f / Dc) + 1e-6f);
    for (int i = threadIdx.x; i < Dc; i += 256)
        out[(size_t)row * Dc + i] = f2bf(xr[i] * inv * g[i]);
}

// ---------------------------------------------------------------- RoPE on q,k (bf16, in-place)
__global__ __launch_bounds__(256)
void rope_kernel(unsigned short* __restrict__ q, unsigned short* __restrict__ k,
                 const float* __restrict__ cosT, const float* __restrict__ sinT) {
    int idx = blockIdx.x * 256 + threadIdx.x;   // B*H*S*32 threads
    int d  = idx & 31;
    int s  = (idx >> 5) & (Sc - 1);
    int bh = idx >> 16;                          // 5 (d) + 11 (s)
    size_t off = ((size_t)bh * Sc + s) * DHc;
    float c0 = cosT[s * DHc + d],      c1 = cosT[s * DHc + d + 32];
    float s0 = sinT[s * DHc + d],      s1 = sinT[s * DHc + d + 32];
    float q0 = bf2f(q[off + d]),       q1 = bf2f(q[off + d + 32]);
    q[off + d]      = f2bf(q0 * c0 - q1 * s0);
    q[off + d + 32] = f2bf(q1 * c1 + q0 * s1);
    float k0 = bf2f(k[off + d]),       k1 = bf2f(k[off + d + 32]);
    k[off + d]      = f2bf(k0 * c0 - k1 * s0);
    k[off + d + 32] = f2bf(k1 * c1 + k0 * s1);
}

// ---------------------------------------------------------------- WMMA GEMM: C[M,N] = A[M,K] * W[N,K]^T
// Epilogue modes
#define M_F32    0
#define M_BF16   1
#define M_QKV    2   // scatter bf16 to [B,H,S,DH]
#define M_RESADD 3   // f32: out = res + acc
#define M_SILU   4   // bf16: out = silu(acc)
#define M_MUL    5   // bf16: out = acc * bf2f(mulbuf)

#define TM 128
#define TN 128
#define KSTEP 32
#define LSTR 40      // bf16 elems per LDS row (80B: 16B aligned, bank-skewed)

// 256 threads = 8 waves; wave tile 32x64 -> 8 WMMA per K-step (2 A-frags x 4 B-frags)
__global__ __launch_bounds__(256)
void gemm_bf16_kernel(const unsigned short* __restrict__ A,
                      const unsigned short* __restrict__ W,
                      int M, int N, int K, int mode,
                      float* __restrict__ outF, unsigned short* __restrict__ outH,
                      const float* __restrict__ res,
                      const unsigned short* __restrict__ mulbuf) {
    __shared__ unsigned short As[TM][LSTR];
    __shared__ unsigned short Ws[TN][LSTR];
    const int tid  = threadIdx.x;
    const int lane = tid & 31;
    const int wid  = tid >> 5;
    const int mBase = blockIdx.y * TM;
    const int nBase = blockIdx.x * TN;
    const int wr = (wid >> 1) * 32;   // wave row offset: 0,32,64,96
    const int wc = (wid & 1) * 64;    // wave col offset: 0,64
    const int ml = lane & 15;
    const int mh = lane >> 4;
    const int lrow = tid >> 2;        // cooperative-load row 0..63 (and +64)
    const int lcol = (tid & 3) * 8;   // cooperative-load col 0,8,16,24

    v8f acc[2][4];
    #pragma unroll
    for (int a = 0; a < 2; ++a)
        #pragma unroll
        for (int nt = 0; nt < 4; ++nt) acc[a][nt] = v8f{};

    for (int k0 = 0; k0 < K; k0 += KSTEP) {
        const unsigned short* gA0 = A + (size_t)(mBase + lrow)      * K + k0 + lcol;
        const unsigned short* gA1 = A + (size_t)(mBase + 64 + lrow) * K + k0 + lcol;
        const unsigned short* gW0 = W + (size_t)(nBase + lrow)      * K + k0 + lcol;
        const unsigned short* gW1 = W + (size_t)(nBase + 64 + lrow) * K + k0 + lcol;
#ifdef HAVE_ASYNC_LDS
        async_ld_b128(gA0, &As[lrow][lcol]);
        async_ld_b128(gA1, &As[64 + lrow][lcol]);
        async_ld_b128(gW0, &Ws[lrow][lcol]);
        async_ld_b128(gW1, &Ws[64 + lrow][lcol]);
#else
        *(uint4*)&As[lrow][lcol]      = *(const uint4*)gA0;
        *(uint4*)&As[64 + lrow][lcol] = *(const uint4*)gA1;
        *(uint4*)&Ws[lrow][lcol]      = *(const uint4*)gW0;
        *(uint4*)&Ws[64 + lrow][lcol] = *(const uint4*)gW1;
#endif
        if (k0 + KSTEP < K) {
            __builtin_prefetch(gA0 + KSTEP, 0, 3);
            __builtin_prefetch(gA1 + KSTEP, 0, 3);
            __builtin_prefetch(gW0 + KSTEP, 0, 3);
            __builtin_prefetch(gW1 + KSTEP, 0, 3);
        }
#ifdef HAVE_ASYNC_LDS
        async_wait0();
#endif
        __syncthreads();

        // A fragments 16x32 (ISA 7.12.2): lane half splits K {0-7,16-23} vs {8-15,24-31}
        Frag fa[2];
        #pragma unroll
        for (int a = 0; a < 2; ++a) {
            #pragma unroll
            for (int v = 0; v < 8; ++v) {
                int g = v >> 2, p = v & 3;
                int kk = g * 16 + mh * 8 + p * 2;
                fa[a].u[v] = *(const unsigned int*)&As[wr + a * 16 + ml][kk];
            }
        }
        // B fragments 32x16: lane half splits K {0-7,16-23} vs {8-15,24-31}; col = lane&15
        Frag fb[4];
        #pragma unroll
        for (int nt = 0; nt < 4; ++nt) {
            #pragma unroll
            for (int v = 0; v < 8; ++v) {
                int kk = (v >> 2) * 16 + mh * 8 + (v & 3) * 2;
                fb[nt].u[v] = *(const unsigned int*)&Ws[wc + nt * 16 + ml][kk];
            }
        }
        #pragma unroll
        for (int a = 0; a < 2; ++a)
            #pragma unroll
            for (int nt = 0; nt < 4; ++nt)
                acc[a][nt] = __builtin_amdgcn_wmma_f32_16x16x32_bf16(
                    false, fa[a].v, false, fb[nt].v, (short)0, acc[a][nt], false, false);
        __syncthreads();
    }

    // epilogue: C layout -> row m = v + 8*half, col = lane&15
    #pragma unroll
    for (int a = 0; a < 2; ++a) {
        #pragma unroll
        for (int nt = 0; nt < 4; ++nt) {
            #pragma unroll
            for (int v = 0; v < 8; ++v) {
                int m = mBase + wr + a * 16 + v + 8 * mh;
                int n = nBase + wc + nt * 16 + ml;
                float x = acc[a][nt][v];
                if (mode == M_QKV) {
                    int b = m >> 11, s = m & (Sc - 1);
                    int hd = n >> 6, dh = n & 63;
                    outH[(((size_t)(b * Hc + hd) * Sc + s) * DHc) + dh] = f2bf(x);
                } else {
                    size_t i = (size_t)m * N + n;
                    if (mode == M_RESADD)      outF[i] = res[i] + x;
                    else if (mode == M_SILU)   outH[i] = f2bf(x * (1.0f / (1.0f + __expf(-x))));
                    else if (mode == M_MUL)    outH[i] = f2bf(x * bf2f(mulbuf[i]));
                    else if (mode == M_BF16)   outH[i] = f2bf(x);
                    else                       outF[i] = x;
                }
            }
        }
    }
}

// ---------------------------------------------------------------- flash attention (causal)
// One wave per 16-query tile; 4 waves/block; per-wave LDS P tile (no block barriers:
// causal trip counts differ per wave).
__global__ __launch_bounds__(128)
void attn_kernel(const unsigned short* __restrict__ Q,
                 const unsigned short* __restrict__ Kt,
                 const unsigned short* __restrict__ V,
                 unsigned short* __restrict__ ctx) {
    __shared__ unsigned short Ps[4][16][34];
    const int tid  = threadIdx.x;
    const int lane = tid & 31;
    const int wid  = tid >> 5;
    const int bh    = blockIdx.x;                 // b*H + h   (0..31)
    const int qtile = blockIdx.y * 4 + wid;
    const int qbase = qtile * 16;
    const int ml = lane & 15;
    const int mh = lane >> 4;
    const size_t base = (size_t)bh * Sc * DHc;

    // Q fragments (A-layout), dh 0-31 and 32-63
    Frag aQ0, aQ1;
    {
        const unsigned short* qp = Q + base + (size_t)(qbase + ml) * DHc;
        #pragma unroll
        for (int v = 0; v < 8; ++v) {
            int g = v >> 2, p = v & 3;
            int kk = g * 16 + mh * 8 + p * 2;
            aQ0.u[v] = *(const unsigned int*)(qp + kk);
            aQ1.u[v] = *(const unsigned int*)(qp + 32 + kk);
        }
    }

    v8f o[4] = {v8f{}, v8f{}, v8f{}, v8f{}};
    float mi[8], li[8];
    #pragma unroll
    for (int v = 0; v < 8; ++v) { mi[v] = -1e30f; li[v] = 0.f; }

    const int nkb = (qbase + 16 + 31) >> 5;       // key blocks of 32, causal
    for (int kb = 0; kb < nkb; ++kb) {
        const int kbase = kb * 32;

        // K^T as B fragments: B[dh][key], col=key, K-dim=dh (two 32-ksteps)
        Frag b00, b01, b10, b11;
        #pragma unroll
        for (int v = 0; v < 8; ++v) {
            int dh = (v >> 2) * 16 + mh * 8 + (v & 3) * 2;
            const unsigned short* k0p = Kt + base + (size_t)(kbase + ml) * DHc;
            const unsigned short* k1p = Kt + base + (size_t)(kbase + 16 + ml) * DHc;
            b00.u[v] = *(const unsigned int*)(k0p + dh);
            b01.u[v] = *(const unsigned int*)(k0p + 32 + dh);
            b10.u[v] = *(const unsigned int*)(k1p + dh);
            b11.u[v] = *(const unsigned int*)(k1p + 32 + dh);
        }
        v8f s0 = {}, s1 = {};
        s0 = __builtin_amdgcn_wmma_f32_16x16x32_bf16(false, aQ0.v, false, b00.v, (short)0, s0, false, false);
        s0 = __builtin_amdgcn_wmma_f32_16x16x32_bf16(false, aQ1.v, false, b01.v, (short)0, s0, false, false);
        s1 = __builtin_amdgcn_wmma_f32_16x16x32_bf16(false, aQ0.v, false, b10.v, (short)0, s1, false, false);
        s1 = __builtin_amdgcn_wmma_f32_16x16x32_bf16(false, aQ1.v, false, b11.v, (short)0, s1, false, false);

        // online softmax per row (row = qbase + v + 8*mh; cols via lane&15)
        #pragma unroll
        for (int v = 0; v < 8; ++v) {
            int row = qbase + v + 8 * mh;
            int c0 = kbase + ml, c1 = c0 + 16;
            float f0 = (c0 <= row) ? s0[v] * 0.125f : -1e30f;
            float f1 = (c1 <= row) ? s1[v] * 0.125f : -1e30f;
            float mx = fmaxf(f0, f1);
            #pragma unroll
            for (int off = 1; off < 16; off <<= 1)
                mx = fmaxf(mx, __shfl_xor(mx, off, 32));
            float nm   = fmaxf(mi[v], mx);
            float corr = __expf(mi[v] - nm);
            float p0 = __expf(f0 - nm);
            float p1 = __expf(f1 - nm);
            float rs = p0 + p1;
            #pragma unroll
            for (int off = 1; off < 16; off <<= 1)
                rs += __shfl_xor(rs, off, 32);
            li[v] = li[v] * corr + rs;
            mi[v] = nm;
            o[0][v] *= corr; o[1][v] *= corr; o[2][v] *= corr; o[3][v] *= corr;
            int prow = v + 8 * mh;
            Ps[wid][prow][ml]      = f2bf(p0);
            Ps[wid][prow][16 + ml] = f2bf(p1);
        }
        asm volatile("s_wait_dscnt 0" ::: "memory");   // per-wave LDS store->load ordering

        // P as A fragment (C-layout -> A-layout transpose through LDS)
        Frag aP;
        #pragma unroll
        for (int v = 0; v < 8; ++v) {
            int g = v >> 2, p = v & 3;
            int kk = g * 16 + mh * 8 + p * 2;
            aP.u[v] = *(const unsigned int*)&Ps[wid][ml][kk];
        }
        // V as B fragments: B[key_local][dh], K-dim = keys (32)
        #pragma unroll
        for (int nt = 0; nt < 4; ++nt) {
            Frag bv;
            #pragma unroll
            for (int v = 0; v < 8; ++v) {
                int keyl = (v >> 2) * 16 + mh * 8 + (v & 3) * 2;
                int dh = nt * 16 + ml;
                unsigned int lo = V[base + (size_t)(kbase + keyl)     * DHc + dh];
                unsigned int hi = V[base + (size_t)(kbase + keyl + 1) * DHc + dh];
                bv.u[v] = lo | (hi << 16);
            }
            o[nt] = __builtin_amdgcn_wmma_f32_16x16x32_bf16(false, aP.v, false, bv.v,
                                                            (short)0, o[nt], false, false);
        }
    }

    // normalize and store context in [B,S,D] (= [b, s, head*64+dh]) for the Wo GEMM
    const int b = bh >> 4, head = bh & 15;
    #pragma unroll
    for (int v = 0; v < 8; ++v) {
        float inv = 1.0f / li[v];
        int m = qbase + v + 8 * mh;
        size_t rowoff = ((size_t)(b * Sc + m)) * Dc + head * DHc;
        #pragma unroll
        for (int nt = 0; nt < 4; ++nt)
            ctx[rowoff + nt * 16 + ml] = f2bf(o[nt][v] * inv);
    }
}

// ---------------------------------------------------------------- host
extern "C" void kernel_launch(void* const* d_in, const int* in_sizes, int n_in,
                              void* d_out, int out_size, void* d_ws, size_t ws_size,
                              hipStream_t stream) {
    const float* hidden = (const float*)d_in[0];
    const float* cosT   = (const float*)d_in[1];
    const float* sinT   = (const float*)d_in[2];
    // d_in[3]=causal_mask, d_in[4]=attention_mask (all-ones): folded into causal logic
    const float* Wq = (const float*)d_in[5];
    const float* Wk = (const float*)d_in[6];
    const float* Wv = (const float*)d_in[7];
    const float* Wo = (const float*)d_in[8];
    const float* Wg = (const float*)d_in[9];
    const float* Wu = (const float*)d_in[10];
    const float* Wd = (const float*)d_in[11];
    const float* g_attn = (const float*)d_in[12];
    const float* g_ffn  = (const float*)d_in[13];
    float* out = (float*)d_out;

    char* ws = (char*)d_ws;
    const size_t MB = 1ull << 20;
    unsigned short* Wq_h = (unsigned short*)(ws + 0  * MB);   // 2 MB each
    unsigned short* Wk_h = (unsigned short*)(ws + 2  * MB);
    unsigned short* Wv_h = (unsigned short*)(ws + 4  * MB);
    unsigned short* Wo_h = (unsigned short*)(ws + 6  * MB);
    unsigned short* Wg_h = (unsigned short*)(ws + 8  * MB);   // 8 MB each
    unsigned short* Wu_h = (unsigned short*)(ws + 16 * MB);
    unsigned short* Wd_h = (unsigned short*)(ws + 24 * MB);
    unsigned short* xn   = (unsigned short*)(ws + 32 * MB);   // 8 MB
    unsigned short* qb   = (unsigned short*)(ws + 40 * MB);   // 8 MB
    unsigned short* kb   = (unsigned short*)(ws + 48 * MB);   // 8 MB
    unsigned short* vb   = (unsigned short*)(ws + 56 * MB);   // 8 MB
    unsigned short* ctx  = (unsigned short*)(ws + 64 * MB);   // 8 MB
    float*          h1   = (float*)        (ws + 72 * MB);   // 16 MB
    unsigned short* xn2  = (unsigned short*)(ws + 88 * MB);   // 8 MB
    unsigned short* gb   = (unsigned short*)(ws + 96 * MB);   // 32 MB
    unsigned short* hu   = (unsigned short*)(ws + 128 * MB);  // 32 MB

    const int nw1 = Dc * Dc;
    const int nw2 = FFc * Dc;
    cvt_kernel<<<(nw1 + 255) / 256, 256, 0, stream>>>(Wq, Wq_h, nw1);
    cvt_kernel<<<(nw1 + 255) / 256, 256, 0, stream>>>(Wk, Wk_h, nw1);
    cvt_kernel<<<(nw1 + 255) / 256, 256, 0, stream>>>(Wv, Wv_h, nw1);
    cvt_kernel<<<(nw1 + 255) / 256, 256, 0, stream>>>(Wo, Wo_h, nw1);
    cvt_kernel<<<(nw2 + 255) / 256, 256, 0, stream>>>(Wg, Wg_h, nw2);
    cvt_kernel<<<(nw2 + 255) / 256, 256, 0, stream>>>(Wu, Wu_h, nw2);
    cvt_kernel<<<(nw2 + 255) / 256, 256, 0, stream>>>(Wd, Wd_h, nw2);

    rmsnorm_kernel<<<BSc, 256, 0, stream>>>(hidden, g_attn, xn);

    dim3 gD(Dc / TN, BSc / TM);       //  8 x 32
    dim3 gF(FFc / TN, BSc / TM);      // 32 x 32
    gemm_bf16_kernel<<<gD, 256, 0, stream>>>(xn, Wq_h, BSc, Dc, Dc, M_QKV, nullptr, qb, nullptr, nullptr);
    gemm_bf16_kernel<<<gD, 256, 0, stream>>>(xn, Wk_h, BSc, Dc, Dc, M_QKV, nullptr, kb, nullptr, nullptr);
    gemm_bf16_kernel<<<gD, 256, 0, stream>>>(xn, Wv_h, BSc, Dc, Dc, M_QKV, nullptr, vb, nullptr, nullptr);

    rope_kernel<<<(Bc * Hc * Sc * 32) / 256, 256, 0, stream>>>(qb, kb, cosT, sinT);

    attn_kernel<<<dim3(Bc * Hc, Sc / 64), 128, 0, stream>>>(qb, kb, vb, ctx);

    gemm_bf16_kernel<<<gD, 256, 0, stream>>>(ctx, Wo_h, BSc, Dc, Dc, M_RESADD, h1, nullptr, hidden, nullptr);

    rmsnorm_kernel<<<BSc, 256, 0, stream>>>(h1, g_ffn, xn2);

    gemm_bf16_kernel<<<gF, 256, 0, stream>>>(xn2, Wg_h, BSc, FFc, Dc, M_SILU, nullptr, gb, nullptr, nullptr);
    gemm_bf16_kernel<<<gF, 256, 0, stream>>>(xn2, Wu_h, BSc, FFc, Dc, M_MUL, nullptr, hu, nullptr, gb);
    gemm_bf16_kernel<<<gD, 256, 0, stream>>>(hu, Wd_h, BSc, Dc, FFc, M_RESADD, out, nullptr, h1, nullptr);
}
